// MultiHeadAttention_14920716387139
// MI455X (gfx1250) — compile-verified, matrix operand order checked by reading, not run on hardware
//
#include <hip/hip_runtime.h>
#include <hip/hip_bf16.h>

typedef __attribute__((ext_vector_type(16))) _Float16 v16h;
typedef __attribute__((ext_vector_type(8)))  _Float16 v8h;
typedef __attribute__((ext_vector_type(8)))  float    v8f;
typedef __attribute__((ext_vector_type(4)))  int      v4i;

#define DEVI __device__ __forceinline__

constexpr int H  = 12;
constexpr int DH = 64;
constexpr int B  = 8;
constexpr int LQ = 1024;
constexpr int LK = 1024;
constexpr int QD = 768;
constexpr int AHD = 768;
constexpr float SCALE = 0.03608439182435161f; // 1/sqrt(768), both stages

DEVI v8f wmma16(v16h a, v16h b, v8f c) {
  return __builtin_amdgcn_wmma_f32_16x16x32_f16(
      false, a, false, b, (short)0, c, false, false);
}

// ---- CDNA5 async global->LDS copy (16B per lane), ASYNCcnt tracked ----------
DEVI void async_cp_b128(void* lds_dst, const void* gsrc) {
#if __has_builtin(__builtin_amdgcn_global_load_async_to_lds_b128)
  __builtin_amdgcn_global_load_async_to_lds_b128(
      (__attribute__((address_space(1))) v4i*)(uintptr_t)gsrc,
      (__attribute__((address_space(3))) v4i*)(uintptr_t)lds_dst, 0, 0);
#else
  asm volatile("global_load_async_to_lds_b128 %0, %1, off"
               :: "v"((unsigned)(uintptr_t)lds_dst), "v"(gsrc) : "memory");
#endif
}

template <int N>
DEVI void wait_async() {  // N is an ICE for the builtin
#if __has_builtin(__builtin_amdgcn_s_wait_asynccnt)
  __builtin_amdgcn_s_wait_asynccnt(N);
#else
  if (N == 0) asm volatile("s_wait_asynccnt 0x0" ::: "memory");
  else if (N == 1) asm volatile("s_wait_asynccnt 0x1" ::: "memory");
  else asm volatile("s_wait_asynccnt 0x2" ::: "memory");
#endif
}

DEVI void sched_fence() {  // per-wave LDS is HW in-order; just stop compiler motion
#if __has_builtin(__builtin_amdgcn_sched_barrier)
  __builtin_amdgcn_sched_barrier(0);
#endif
  asm volatile("" ::: "memory");
}

// A-fragment (16x32 f16): lane = row, halves 0-7 -> K=hi*8..+7, 8-15 -> K=16+hi*8..+7
// Also the B-fragment of X^T products: B(k,n)=X[n][k] with lane = n.
DEVI v16h frag_h(const _Float16* __restrict__ src, int ld) {
  const int lane = threadIdx.x & 31;
  const int r  = lane & 15;
  const int hi = lane >> 4;
  const v8h x0 = *(const v8h*)(src + (size_t)r * ld + hi * 8);
  const v8h x1 = *(const v8h*)(src + (size_t)r * ld + 16 + hi * 8);
  v16h a;
#pragma unroll
  for (int i = 0; i < 8; ++i) { a[i] = x0[i]; a[i + 8] = x1[i]; }
  return a;
}

DEVI v16h frag_f32(const float* __restrict__ src, int ld) {
  const int lane = threadIdx.x & 31;
  const int r  = lane & 15;
  const int hi = lane >> 4;
  const float4 x0 = *(const float4*)(src + (size_t)r * ld + hi * 8);
  const float4 x1 = *(const float4*)(src + (size_t)r * ld + hi * 8 + 4);
  const float4 x2 = *(const float4*)(src + (size_t)r * ld + 16 + hi * 8);
  const float4 x3 = *(const float4*)(src + (size_t)r * ld + 16 + hi * 8 + 4);
  v16h a;
  a[0]  = (_Float16)x0.x; a[1]  = (_Float16)x0.y; a[2]  = (_Float16)x0.z; a[3]  = (_Float16)x0.w;
  a[4]  = (_Float16)x1.x; a[5]  = (_Float16)x1.y; a[6]  = (_Float16)x1.z; a[7]  = (_Float16)x1.w;
  a[8]  = (_Float16)x2.x; a[9]  = (_Float16)x2.y; a[10] = (_Float16)x2.z; a[11] = (_Float16)x2.w;
  a[12] = (_Float16)x3.x; a[13] = (_Float16)x3.y; a[14] = (_Float16)x3.z; a[15] = (_Float16)x3.w;
  return a;
}

// ---------------- Kernel 1: fused QKV projection + head split + f16 convert ----
__global__ __launch_bounds__(256) void k_proj(
    const float* __restrict__ query, const float* __restrict__ key_,
    const float* __restrict__ Wq, const float* __restrict__ bq,
    const float* __restrict__ Wk, const float* __restrict__ bk,
    const float* __restrict__ Wv, const float* __restrict__ bv,
    _Float16* __restrict__ q16, _Float16* __restrict__ k16,
    _Float16* __restrict__ vT) {
  const int TPM = (B * LQ) / 16;  // 512 row tiles
  const int TPN = AHD / 16;       // 48 col tiles
  const int wid = blockIdx.x * 8 + (threadIdx.x >> 5);
  const int mat = wid / (TPM * TPN);
  const int rem = wid % (TPM * TPN);
  const int m0 = (rem / TPN) * 16;
  const int n0 = (rem % TPN) * 16;
  const int lane = threadIdx.x & 31;
  const int nn = lane & 15, hi = lane >> 4;

  const float* X    = (mat == 0) ? query : key_;
  const float* W    = (mat == 0) ? Wq : (mat == 1 ? Wk : Wv);
  const float* bias = (mat == 0) ? bq : (mat == 1 ? bk : bv);

  v8f acc;
  const float b0 = bias[n0 + nn];
#pragma unroll
  for (int j = 0; j < 8; ++j) acc[j] = b0;

  for (int kk = 0; kk < QD; kk += 32) {
    if (kk + 64 < QD) {
      __builtin_prefetch(X + (size_t)m0 * QD + kk + 64, 0, 3);
      __builtin_prefetch(W + (size_t)n0 * QD + kk + 64, 0, 3);
    }
    const v16h a = frag_f32(X + (size_t)m0 * QD + kk, QD);
    const v16h w = frag_f32(W + (size_t)n0 * QD + kk, QD);  // B(k,n)=W[n][k]
    acc = wmma16(a, w, acc);
  }

  const int hh = n0 >> 6, d0 = n0 & 63;
  const int bb = m0 >> 10, l0 = m0 & 1023;
  if (mat == 2) {
    v8h pk;
#pragma unroll
    for (int j = 0; j < 8; ++j) pk[j] = (_Float16)acc[j];
    *(v8h*)(vT + ((size_t)((hh * B + bb) * DH + d0 + nn)) * LK + l0 + 8 * hi) = pk;
  } else {
    _Float16* dst = (mat == 0) ? q16 : k16;
#pragma unroll
    for (int j = 0; j < 8; ++j) {
      const int l = l0 + j + 8 * hi;
      dst[((size_t)((hh * B + bb) * LQ + l)) * DH + d0 + nn] = (_Float16)acc[j];
    }
  }
}

// ---------------- Kernel 2: stage-1 flash attention -> o1 (and o1^T) ----------
// Block = 8 waves on same (h,b); K/V tiles staged once per block via async DMA.
__global__ __launch_bounds__(256) void k_attn1(
    const _Float16* __restrict__ q16, const _Float16* __restrict__ k16,
    const _Float16* __restrict__ vT, const int* __restrict__ key_mask,
    _Float16* __restrict__ o1, _Float16* __restrict__ o1T) {
  __shared__ __align__(16) _Float16 Kt[2][32 * 64];  // [keys][d]
  __shared__ __align__(16) _Float16 Vt[2][64 * 32];  // [d][keys]
  __shared__ __align__(16) _Float16 Pbuf[8][16 * 32];
  const int tid = threadIdx.x;
  const int w = tid >> 5;
  const int wid = blockIdx.x * 8 + w;
  const int h   = wid / (B * (LQ / 16));
  const int rem = wid % (B * (LQ / 16));
  const int b   = rem / (LQ / 16);
  const int q0  = (rem % (LQ / 16)) * 16;
  const int lane = tid & 31;
  const int nn = lane & 15, hi = lane >> 4;

  const _Float16* Qb = q16 + ((size_t)((h * B + b) * LQ + q0)) * DH;
  const _Float16* Kb = k16 + ((size_t)(h * B + b)) * LK * DH;
  const _Float16* Vb = vT  + ((size_t)(h * B + b)) * DH * LK;
  const int* km = key_mask + b * LK;

  const int vr = tid >> 2, vc = (tid & 3) * 8;  // V-tile copy coords
  auto issue = [&](int kt, int buf) {  // 2 x 16B per thread = 2 x 4KB per block
    async_cp_b128(&Kt[buf][tid * 8], Kb + (size_t)kt * DH + tid * 8);
    async_cp_b128(&Vt[buf][vr * 32 + vc], Vb + (size_t)vr * LK + kt + vc);
  };

  const v16h aq0 = frag_h(Qb, DH);
  const v16h aq1 = frag_h(Qb + 32, DH);

  v8f acc[4];
  const v8f z = {};
#pragma unroll
  for (int t = 0; t < 4; ++t) acc[t] = z;
  float rm[8], rl[8];
#pragma unroll
  for (int j = 0; j < 8; ++j) { rm[j] = -1e30f; rl[j] = 0.f; }

  issue(0, 0);
  for (int i = 0; i < LK / 32; ++i) {
    const int kt = i * 32, buf = i & 1;
    if (i + 1 < LK / 32) { issue(kt + 32, buf ^ 1); wait_async<2>(); }
    else                 { wait_async<0>(); }
    __syncthreads();  // tile i resident for all waves

    v8f s0 = wmma16(aq0, frag_h(&Kt[buf][0], DH), z);
    s0     = wmma16(aq1, frag_h(&Kt[buf][32], DH), s0);
    v8f s1 = wmma16(aq0, frag_h(&Kt[buf][16 * DH], DH), z);
    s1     = wmma16(aq1, frag_h(&Kt[buf][16 * DH + 32], DH), s1);

    const bool ok0 = km[kt + nn] == 1;
    const bool ok1 = km[kt + 16 + nn] == 1;
    float alpha[8];
#pragma unroll
    for (int j = 0; j < 8; ++j) {
      const float v0 = ok0 ? s0[j] * SCALE : -1e30f;
      const float v1 = ok1 ? s1[j] * SCALE : -1e30f;
      float cm = fmaxf(v0, v1);
#pragma unroll
      for (int x = 1; x < 16; x <<= 1) cm = fmaxf(cm, __shfl_xor(cm, x, 32));
      const float mn = fmaxf(rm[j], cm);
      const float al = __expf(rm[j] - mn);
      const float p0 = __expf(v0 - mn);
      const float p1 = __expf(v1 - mn);
      float ps = p0 + p1;
#pragma unroll
      for (int x = 1; x < 16; x <<= 1) ps += __shfl_xor(ps, x, 32);
      rl[j] = rl[j] * al + ps;
      rm[j] = mn;
      alpha[j] = al;
      Pbuf[w][(j + 8 * hi) * 32 + nn]      = (_Float16)p0;
      Pbuf[w][(j + 8 * hi) * 32 + 16 + nn] = (_Float16)p1;
    }
#pragma unroll
    for (int t = 0; t < 4; ++t)
#pragma unroll
      for (int j = 0; j < 8; ++j) acc[t][j] *= alpha[j];
    sched_fence();  // wave-private Pbuf: HW keeps per-wave LDS in order
    const v16h ap = frag_h(&Pbuf[w][0], 32);
#pragma unroll
    for (int t = 0; t < 4; ++t)
      acc[t] = wmma16(ap, frag_h(&Vt[buf][(t * 16) * 32], 32), acc[t]);
    __syncthreads();  // all waves done with buf before it is re-filled
  }

  float il[8];
#pragma unroll
  for (int j = 0; j < 8; ++j) il[j] = 1.0f / rl[j];
  _Float16* o1b  = o1  + ((size_t)((h * B + b) * LQ + q0)) * DH;
  _Float16* o1Tb = o1T + ((size_t)(h * B + b)) * DH * LK;
#pragma unroll
  for (int t = 0; t < 4; ++t) {
    v8h pk;
#pragma unroll
    for (int j = 0; j < 8; ++j) {
      const float v = acc[t][j] * il[j];
      pk[j] = (_Float16)v;
      o1b[(size_t)(j + 8 * hi) * DH + t * 16 + nn] = (_Float16)v;
    }
    *(v8h*)(o1Tb + (size_t)(t * 16 + nn) * LK + q0 + 8 * hi) = pk;
  }
}

// ---------------- Kernel 3: stage-2 softmax row stats --------------------------
__global__ __launch_bounds__(256) void k_stats2(
    const _Float16* __restrict__ o1, const int* __restrict__ query_mask,
    float* __restrict__ mstat, float* __restrict__ lstat) {
  __shared__ __align__(16) _Float16 Ot[2][32 * 64];
  const int tid = threadIdx.x;
  const int wid = blockIdx.x * 8 + (tid >> 5);
  const int h   = wid / (B * (LQ / 16));
  const int rem = wid % (B * (LQ / 16));
  const int b   = rem / (LQ / 16);
  const int q0  = (rem % (LQ / 16)) * 16;
  const int lane = tid & 31;
  const int nn = lane & 15, hi = lane >> 4;

  const _Float16* Ob = o1 + ((size_t)(h * B + b)) * LQ * DH;
  const int* qm = query_mask + b * LQ;
  const v16h aq0 = frag_h(Ob + (size_t)q0 * DH, DH);
  const v16h aq1 = frag_h(Ob + (size_t)q0 * DH + 32, DH);
  const v8f z = {};
  float rm[8], rl[8];
#pragma unroll
  for (int j = 0; j < 8; ++j) { rm[j] = -1e30f; rl[j] = 0.f; }

  async_cp_b128(&Ot[0][tid * 8], Ob + tid * 8);
  for (int i = 0; i < LQ / 32; ++i) {
    const int kt = i * 32, buf = i & 1;
    if (i + 1 < LQ / 32) {
      async_cp_b128(&Ot[buf ^ 1][tid * 8], Ob + (size_t)(kt + 32) * DH + tid * 8);
      wait_async<1>();
    } else {
      wait_async<0>();
    }
    __syncthreads();
    v8f s0 = wmma16(aq0, frag_h(&Ot[buf][0], DH), z);
    s0     = wmma16(aq1, frag_h(&Ot[buf][32], DH), s0);
    v8f s1 = wmma16(aq0, frag_h(&Ot[buf][16 * DH], DH), z);
    s1     = wmma16(aq1, frag_h(&Ot[buf][16 * DH + 32], DH), s1);
    const bool ok0 = qm[kt + nn] == 1;
    const bool ok1 = qm[kt + 16 + nn] == 1;
#pragma unroll
    for (int j = 0; j < 8; ++j) {
      const float v0 = ok0 ? s0[j] * SCALE : -1e30f;
      const float v1 = ok1 ? s1[j] * SCALE : -1e30f;
      float cm = fmaxf(v0, v1);
#pragma unroll
      for (int x = 1; x < 16; x <<= 1) cm = fmaxf(cm, __shfl_xor(cm, x, 32));
      const float mn = fmaxf(rm[j], cm);
      const float al = __expf(rm[j] - mn);
      float ps = __expf(v0 - mn) + __expf(v1 - mn);
#pragma unroll
      for (int x = 1; x < 16; x <<= 1) ps += __shfl_xor(ps, x, 32);
      rl[j] = rl[j] * al + ps;
      rm[j] = mn;
    }
    __syncthreads();
  }
  if (nn == 0) {
#pragma unroll
    for (int j = 0; j < 8; ++j) {
      const size_t idx = (size_t)(h * B + b) * LQ + q0 + j + 8 * hi;
      mstat[idx] = rm[j];
      lstat[idx] = rl[j];
    }
  }
}

// ---------------- Kernel 4: emit normalized p2 once + o2 = p2*o1 + tanh -------
__global__ __launch_bounds__(256) void k_attn2(
    const _Float16* __restrict__ o1, const _Float16* __restrict__ o1T,
    const int* __restrict__ query_mask,
    const float* __restrict__ mstat, const float* __restrict__ lstat,
    float* __restrict__ out, float* __restrict__ p2) {
  __shared__ __align__(16) _Float16 Ot[2][32 * 64];   // o1 tile [keys][d]
  __shared__ __align__(16) _Float16 OTt[2][64 * 32];  // o1^T tile [d][keys]
  __shared__ __align__(16) _Float16 Pbuf[8][16 * 32];
  const int tid = threadIdx.x;
  const int w = tid >> 5;
  const int wid = blockIdx.x * 8 + w;
  const int h   = wid / (B * (LQ / 16));
  const int rem = wid % (B * (LQ / 16));
  const int b   = rem / (LQ / 16);
  const int q0  = (rem % (LQ / 16)) * 16;
  const int lane = tid & 31;
  const int nn = lane & 15, hi = lane >> 4;

  const _Float16* Ob  = o1  + ((size_t)(h * B + b)) * LQ * DH;
  const _Float16* OTb = o1T + ((size_t)(h * B + b)) * DH * LK;
  const int* qm = query_mask + b * LQ;
  const v16h aq0 = frag_h(Ob + (size_t)q0 * DH, DH);
  const v16h aq1 = frag_h(Ob + (size_t)q0 * DH + 32, DH);
  const v8f z = {};

  const int vr = tid >> 2, vc = (tid & 3) * 8;
  auto issue = [&](int kt, int buf) {
    async_cp_b128(&Ot[buf][tid * 8], Ob + (size_t)kt * DH + tid * 8);
    async_cp_b128(&OTt[buf][vr * 32 + vc], OTb + (size_t)vr * LK + kt + vc);
  };

  float m_[8], il[8];
#pragma unroll
  for (int j = 0; j < 8; ++j) {
    const size_t idx = (size_t)(h * B + b) * LQ + q0 + j + 8 * hi;
    m_[j] = mstat[idx];
    il[j] = 1.0f / lstat[idx];
  }
  float* p2b = p2 + ((size_t)(h * B + b) * LQ + q0) * LK;
  v8f acc[4];
#pragma unroll
  for (int t = 0; t < 4; ++t) acc[t] = z;

  issue(0, 0);
  for (int i = 0; i < LQ / 32; ++i) {
    const int kt = i * 32, buf = i & 1;
    if (i + 1 < LQ / 32) { issue(kt + 32, buf ^ 1); wait_async<2>(); }
    else                 { wait_async<0>(); }
    __syncthreads();

    v8f s0 = wmma16(aq0, frag_h(&Ot[buf][0], DH), z);
    s0     = wmma16(aq1, frag_h(&Ot[buf][32], DH), s0);
    v8f s1 = wmma16(aq0, frag_h(&Ot[buf][16 * DH], DH), z);
    s1     = wmma16(aq1, frag_h(&Ot[buf][16 * DH + 32], DH), s1);
    const bool ok0 = qm[kt + nn] == 1;
    const bool ok1 = qm[kt + 16 + nn] == 1;
#pragma unroll
    for (int j = 0; j < 8; ++j) {
      const float p0 = ok0 ? __expf(s0[j] * SCALE - m_[j]) * il[j] : 0.f;
      const float p1 = ok1 ? __expf(s1[j] * SCALE - m_[j]) * il[j] : 0.f;
      p2b[(size_t)(j + 8 * hi) * LK + kt + nn]      = p0;  // single pass over 402MB
      p2b[(size_t)(j + 8 * hi) * LK + kt + 16 + nn] = p1;
      Pbuf[w][(j + 8 * hi) * 32 + nn]      = (_Float16)p0;
      Pbuf[w][(j + 8 * hi) * 32 + 16 + nn] = (_Float16)p1;
    }
    sched_fence();
    const v16h ap = frag_h(&Pbuf[w][0], 32);
#pragma unroll
    for (int t = 0; t < 4; ++t)
      acc[t] = wmma16(ap, frag_h(&OTt[buf][(t * 16) * 32], 32), acc[t]);
    __syncthreads();
  }

#pragma unroll
  for (int t = 0; t < 4; ++t)
#pragma unroll
    for (int j = 0; j < 8; ++j) {
      out[((size_t)(b * LQ + q0 + j + 8 * hi)) * AHD + h * DH + t * 16 + nn] =
          tanhf(acc[t][j]);
    }
}

extern "C" void kernel_launch(void* const* d_in, const int* in_sizes, int n_in,
                              void* d_out, int out_size, void* d_ws, size_t ws_size,
                              hipStream_t stream) {
  (void)in_sizes; (void)n_in; (void)out_size; (void)ws_size;
  const float* query = (const float*)d_in[0];
  const float* key_  = (const float*)d_in[1];
  const int*   qmask = (const int*)d_in[2];
  const int*   kmask = (const int*)d_in[3];
  const float* Wq = (const float*)d_in[4];
  const float* bq = (const float*)d_in[5];
  const float* Wk = (const float*)d_in[6];
  const float* bk = (const float*)d_in[7];
  const float* Wv = (const float*)d_in[8];
  const float* bv = (const float*)d_in[9];

  _Float16* ws = (_Float16*)d_ws;
  const size_t SZ = (size_t)H * B * LQ * DH;   // 6,291,456 halves per tensor
  _Float16* q16 = ws;
  _Float16* k16 = ws + SZ;
  _Float16* vT  = ws + 2 * SZ;
  _Float16* o1  = ws + 3 * SZ;
  _Float16* o1T = ws + 4 * SZ;
  float* mstat = (float*)(ws + 5 * SZ);
  float* lstat = mstat + (size_t)H * B * LQ;

  float* out = (float*)d_out;
  float* p2  = out + (size_t)B * LQ * AHD;

  k_proj<<<dim3(9216), dim3(256), 0, stream>>>(query, key_, Wq, bq, Wk, bk, Wv, bv,
                                               q16, k16, vT);
  k_attn1<<<dim3(768), dim3(256), 0, stream>>>(q16, k16, vT, kmask, o1, o1T);
  k_stats2<<<dim3(768), dim3(256), 0, stream>>>(o1, qmask, mstat, lstat);
  k_attn2<<<dim3(768), dim3(256), 0, stream>>>(o1, o1T, qmask, mstat, lstat, out, p2);
}
